// SSBlock_51153060496032
// MI455X (gfx1250) — matrix operand order repaired and checked by reading, not compiled, not run on hardware
//
#include <hip/hip_runtime.h>
#include <hip/hip_bf16.h>

// ---------------- problem constants ----------------
#define B_    4
#define C_    64
#define H_    256
#define W_    256
#define HW_   65536            // H_*W_
#define NPIX  262144           // B_*HW_
#define HEADS 4
#define DH    16

typedef __attribute__((ext_vector_type(16))) _Float16 v16h;
typedef __attribute__((ext_vector_type(8)))  float    v8f;
typedef __attribute__((ext_vector_type(2)))  __fp16   fp16x2;

union Frag { v16h h; unsigned int w[8]; };

// single-instruction packed convert: v_cvt_pk_rtz_f16_f32
__device__ __forceinline__ unsigned int pk(float a, float b) {
  union { fp16x2 h; unsigned int u; } t;
  t.h = __builtin_amdgcn_cvt_pkrtz(a, b);
  return t.u;
}
__device__ __forceinline__ float2 up(unsigned int u) {
  union { unsigned int u; _Float16 h[2]; } t; t.u = u;
  return make_float2((float)t.h[0], (float)t.h[1]);
}
__device__ __forceinline__ float half_of(unsigned int u, int i) {
  union { unsigned int u; _Float16 h[2]; } t; t.u = u; return (float)t.h[i];
}
__device__ __forceinline__ v8f wmmaf(const Frag& a, const Frag& b, v8f c) {
  return __builtin_amdgcn_wmma_f32_16x16x32_f16(false, a.h, false, b.h,
                                                (short)0, c, false, false);
}

// A-fragment loaders (16-bit A 16x32, this code's contiguous k order: lane m=L%16, hi=L/16;
// reg j -> K = kbase + hi*16 + 2*j, pair packed)
// W stored out-major: A[m][k] = W[m*ldk + k]
__device__ __forceinline__ void loadA_mk(Frag& f, const float* W, int ldk,
                                         int kbase, int lane) {
  int m = lane & 15, hi = lane >> 4;
#pragma unroll
  for (int j = 0; j < 8; ++j) {
    int k0 = kbase + hi * 16 + 2 * j;
    f.w[j] = pk(W[m * ldk + k0], W[m * ldk + k0 + 1]);
  }
}
// W stored in-major (rows = k, stride 64): A[m][k] = W[k*64 + mbase + m]
__device__ __forceinline__ void loadA_km(Frag& f, const float* W, int mbase,
                                         int kbase, int lane) {
  int m = lane & 15, hi = lane >> 4;
#pragma unroll
  for (int j = 0; j < 8; ++j) {
    int k0 = kbase + hi * 16 + 2 * j;
    f.w[j] = pk(W[k0 * 64 + mbase + m], W[(k0 + 1) * 64 + mbase + m]);
  }
}
// spa_c2 OIHW [64][32][3][3]: A[m][k] = W[((mbase+m)*32 + k)*9 + tap]
__device__ __forceinline__ void loadA_c2(Frag& f, const float* W, int mbase,
                                         int tap, int lane) {
  int m = lane & 15, hi = lane >> 4;
#pragma unroll
  for (int j = 0; j < 8; ++j) {
    int k0 = hi * 16 + 2 * j;
    f.w[j] = pk(W[((mbase + m) * 32 + k0) * 9 + tap],
                W[((mbase + m) * 32 + k0 + 1) * 9 + tap]);
  }
}

// ---------------- kernel 0: zero accumulators ----------------
__global__ void zero_kernel(float* p, int n) {
  int i = blockIdx.x * blockDim.x + threadIdx.x;
  if (i < n) p[i] = 0.f;
}

// ---------------- kernel 1: QKV projection GEMM ----------------
// 12 waves: waves 0-3 -> q tiles, 4-7 -> k, 8-11 -> v. Weight-stationary,
// x staged to LDS as f16 channel-pairs. Output pair-packed [cp][pix] f16.
__global__ void __launch_bounds__(384)
qkv_kernel(const float* __restrict__ x, const float* __restrict__ Wq,
           const float* __restrict__ Wk, const float* __restrict__ Wv,
           unsigned int* __restrict__ qws, unsigned int* __restrict__ kws,
           unsigned int* __restrict__ vws) {
  __shared__ unsigned int xs[32 * 256];
  const int t = threadIdx.x;
  const long pix0 = (long)blockIdx.x * 256;  // 1024 chunks of 256 pixels
  for (int e = t; e < 32 * 256; e += 384) {
    int cp = e >> 8, p = e & 255;
    long gp = pix0 + p;
    long b = gp >> 16, hw = gp & (HW_ - 1);
    const float* xb = x + ((b * 64) << 16) + hw;
    xs[e] = pk(xb[(long)(2 * cp) << 16], xb[(long)(2 * cp + 1) << 16]);
  }
  __syncthreads();
  const int wave = t >> 5, lane = t & 31;
  const int n = lane & 15, hi = lane >> 4;
  const float* Wsel = (wave < 4) ? Wq : ((wave < 8) ? Wk : Wv);
  unsigned int* dst = (wave < 4) ? qws : ((wave < 8) ? kws : vws);
  const int ot = wave & 3;
  Frag a0, a1;
  loadA_km(a0, Wsel, ot * 16, 0, lane);
  loadA_km(a1, Wsel, ot * 16, 32, lane);
  for (int pt = 0; pt < 16; ++pt) {
    Frag b0, b1;
#pragma unroll
    for (int j = 0; j < 8; ++j) {
      b0.w[j] = xs[(hi * 8 + j) * 256 + pt * 16 + n];
      b1.w[j] = xs[(16 + hi * 8 + j) * 256 + pt * 16 + n];
    }
    v8f acc = {};
    acc = wmmaf(a0, b0, acc);
    acc = wmmaf(a1, b1, acc);
    long p = pix0 + pt * 16 + n;
#pragma unroll
    for (int r = 0; r < 8; r += 2) {
      int ch = ot * 16 + r + hi * 8;                 // even channel
      dst[(long)(ch >> 1) * NPIX + p] = pk(acc[r], acc[r + 1]);
    }
  }
}

// ---------------- kernel 2: Gram (K Q^T over pixels) + sum-of-squares ----
__global__ void __launch_bounds__(256)
gram_kernel(const unsigned int* __restrict__ qws,
            const unsigned int* __restrict__ kws,
            float* accQ, float* accK, float* G) {
  __shared__ unsigned int qs[32 * 64], ks[32 * 64];
  const int t = threadIdx.x;
  const int b = blockIdx.y;
  const long base = (long)b * HW_ + (long)blockIdx.x * 1024;
  float g0 = 0.f, g1 = 0.f, g2 = 0.f, g3 = 0.f;
  float sq = 0.f;
  for (int st = 0; st < 16; ++st) {
    __syncthreads();
    for (int e = t; e < 32 * 64; e += 256) {
      int cp = e >> 6, p = e & 63;
      long pix = base + st * 64 + p;
      qs[e] = qws[(long)cp * NPIX + pix];
      ks[e] = kws[(long)cp * NPIX + pix];
    }
    __syncthreads();
    float* gp[4] = {&g0, &g1, &g2, &g3};
#pragma unroll
    for (int i = 0; i < 4; ++i) {
      int idx = t * 4 + i;                        // (h,d,e)
      int h = idx >> 8, d = (idx >> 4) & 15, e = idx & 15;
      int ck = h * 16 + d, cq = h * 16 + e;
      float acc = 0.f;
      for (int p = 0; p < 64; ++p) {
        float kv = half_of(ks[(ck >> 1) * 64 + p], ck & 1);
        float qv = half_of(qs[(cq >> 1) * 64 + p], cq & 1);
        acc += kv * qv;
      }
      *gp[i] += acc;
    }
    if (t < 128) {
      int c = t & 63;
      const unsigned int* s = (t < 64) ? qs : ks;
      for (int p = 0; p < 64; ++p) {
        float v = half_of(s[(c >> 1) * 64 + p], c & 1);
        sq += v * v;
      }
    }
  }
  atomicAdd(&G[b * 1024 + t * 4 + 0], g0);
  atomicAdd(&G[b * 1024 + t * 4 + 1], g1);
  atomicAdd(&G[b * 1024 + t * 4 + 2], g2);
  atomicAdd(&G[b * 1024 + t * 4 + 3], g3);
  if (t < 64)       atomicAdd(&accQ[b * 64 + t], sq);
  else if (t < 128) atomicAdd(&accK[b * 64 + (t - 64)], sq);
}

// ---------------- kernel 3: softmax + Weff + LN-folded ffn_c1 ----------
__global__ void __launch_bounds__(256)
prep_kernel(const float* accQ, const float* accK, const float* G,
            const float* rescale, const float* Wp,
            const float* ln_w, const float* ln_b,
            const float* fc1w, const float* fc1b,
            float* Weff, float* W1f, float* b1f) {
  __shared__ float A[4 * 4 * 16 * 16];
  const int t = threadIdx.x;
  {  // one softmax row per thread: (b,h,d)
    int b = t >> 6, h = (t >> 4) & 3, d = t & 15;
    float kn = fmaxf(sqrtf(accK[b * 64 + h * 16 + d]), 1e-12f);
    float row[16], mx = -1e30f;
    for (int e = 0; e < 16; ++e) {
      float qn = fmaxf(sqrtf(accQ[b * 64 + h * 16 + e]), 1e-12f);
      float v = G[((b * 4 + h) * 16 + d) * 16 + e] / (kn * qn) * rescale[h];
      row[e] = v; mx = fmaxf(mx, v);
    }
    float s = 0.f;
    for (int e = 0; e < 16; ++e) { row[e] = __expf(row[e] - mx); s += row[e]; }
    for (int e = 0; e < 16; ++e) A[((b * 4 + h) * 16 + d) * 16 + e] = row[e] / s;
  }
  __syncthreads();
  // Weff[b][c][i], i = h*16+e:  sum_d Wp[h*16+d][c] * attn[b][h][d][e]
  for (int e2 = t; e2 < 4 * 64 * 64; e2 += 256) {
    int b = e2 >> 12, c = (e2 >> 6) & 63, i = e2 & 63;
    int h = i >> 4, e = i & 15;
    float s = 0.f;
    for (int d = 0; d < 16; ++d)
      s += Wp[(h * 16 + d) * 64 + c] * A[((b * 4 + h) * 16 + d) * 16 + e];
    Weff[e2] = s;
  }
  for (int e2 = t; e2 < 64 * 64; e2 += 256) {
    int o = e2 >> 6, i = e2 & 63;
    W1f[e2] = fc1w[o * 64 + i] * ln_w[i];
  }
  if (t < 64) {
    float s = fc1b[t];
    for (int i = 0; i < 64; ++i) s += fc1w[t * 64 + i] * ln_b[i];
    b1f[t] = s;
  }
}

// ---------------- kernel 4: fused spatial block ----------------
// 16x16 output tile, halo 2.  Stages:
//  1) x patch -> LDS  2) s1 = 1x1 conv (WMMA)  3) dw3x3 + gate (VALU)
//  4) 3x3 conv 32->64 (9x WMMA) + Weff*v (2x WMMA) -> out  5) LayerNorm
//  6) folded ffn_c1 (WMMA)  7) gate  8) ffn_c2 (WMMA) + residuals -> store
__global__ void __launch_bounds__(256)
fused_kernel(const float* __restrict__ x, const unsigned int* __restrict__ vws,
             const float* __restrict__ Weff, const float* __restrict__ bp,
             const float* __restrict__ p1w, const float* __restrict__ p1b,
             const float* __restrict__ dww, const float* __restrict__ dwb,
             const float* __restrict__ c2w, const float* __restrict__ c2b,
             const float* __restrict__ W1f, const float* __restrict__ b1f,
             const float* __restrict__ fc2w, const float* __restrict__ fc2b,
             float* __restrict__ out) {
  extern __shared__ char smem[];
  unsigned int* XB = (unsigned int*)smem;       // [32][400] x patch (f16 pairs)
  unsigned int* SB = XB + 32 * 400;             // [32][400] s1; reused as OUT [32][256]
  unsigned int* GB = SB + 32 * 400;             // [16][324] gate; reused [16][256]
  unsigned int* HB = GB + 16 * 324;             // [32][256] ffn hidden
  float* MU = (float*)(HB + 32 * 256);          // [256]
  float* RS = MU + 256;                         // [256]

  const int t = threadIdx.x, wave = t >> 5, lane = t & 31;
  const int n = lane & 15, hi = lane >> 4;
  const int tx = blockIdx.x, ty = blockIdx.y, b = blockIdx.z;
  const int gx0 = tx * 16 - 2, gy0 = ty * 16 - 2;
  const float* xb = x + (long)b * 64 * HW_;

  // stage 1: x patch (zero-padded)
  for (int e = t; e < 32 * 400; e += 256) {
    int cp = e / 400, pp = e % 400, py = pp / 20, px = pp % 20;
    int gy = gy0 + py, gx = gx0 + px;
    unsigned int v = 0u;
    if ((unsigned)gy < 256u && (unsigned)gx < 256u) {
      long o = ((long)(2 * cp) << 16) + gy * 256 + gx;
      v = pk(xb[o], xb[o + HW_]);
    }
    XB[e] = v;
  }
  __syncthreads();

  // stage 2: s1 = spa_p1 1x1 conv over 400 patch pixels (25 pixel-tiles)
  {
    const int ot = wave & 3;
    Frag a0, a1;
    loadA_mk(a0, p1w + ot * 16 * 64, 64, 0, lane);
    loadA_mk(a1, p1w + ot * 16 * 64, 64, 32, lane);
    float bias[8];
#pragma unroll
    for (int r = 0; r < 8; ++r) bias[r] = p1b[ot * 16 + r + hi * 8];
    for (int pt = wave >> 2; pt < 25; pt += 2) {
      Frag b0, b1;
#pragma unroll
      for (int j = 0; j < 8; ++j) {
        b0.w[j] = XB[(hi * 8 + j) * 400 + pt * 16 + n];
        b1.w[j] = XB[(16 + hi * 8 + j) * 400 + pt * 16 + n];
      }
      v8f acc = {};
      acc = wmmaf(a0, b0, acc);
      acc = wmmaf(a1, b1, acc);
      int pp = pt * 16 + n, py = pp / 20, px = pp % 20;
      bool valid = ((unsigned)(gy0 + py) < 256u) && ((unsigned)(gx0 + px) < 256u);
#pragma unroll
      for (int r = 0; r < 8; r += 2)
        SB[(ot * 8 + (r >> 1) + hi * 4) * 400 + pp] =
            valid ? pk(acc[r] + bias[r], acc[r + 1] + bias[r + 1]) : 0u;
    }
  }
  __syncthreads();

  // stage 3: depthwise 3x3 + gate (sa*sb) on 18x18 gate patch
  for (int e = t; e < 16 * 324; e += 256) {
    int cp = e / 324, pp = e % 324, oy = pp / 18, ox = pp % 18;
    int gy = gy0 + oy + 1, gx = gx0 + ox + 1;
    if ((unsigned)gy >= 256u || (unsigned)gx >= 256u) { GB[e] = 0u; continue; }
    int cA = 2 * cp, cB = 2 * cp + 32;
    float a0v = dwb[cA], a1v = dwb[cA + 1], b0v = dwb[cB], b1v = dwb[cB + 1];
#pragma unroll
    for (int dy = 0; dy < 3; ++dy)
#pragma unroll
      for (int dx = 0; dx < 3; ++dx) {
        int sp = (oy + dy) * 20 + ox + dx, wi = dy * 3 + dx;
        float2 sA = up(SB[cp * 400 + sp]);
        float2 sB2 = up(SB[(cp + 16) * 400 + sp]);
        a0v += sA.x * dww[cA * 9 + wi];
        a1v += sA.y * dww[(cA + 1) * 9 + wi];
        b0v += sB2.x * dww[cB * 9 + wi];
        b1v += sB2.y * dww[(cB + 1) * 9 + wi];
      }
    GB[e] = pk(a0v * b0v, a1v * b1v);
  }
  __syncthreads();

  // stage 4+5: spa_c2 3x3 (9 taps, K=32) + Weff*v (K=64) + biases -> OUT
  {
    const int ot = wave & 3;
    Frag ac[9];
#pragma unroll
    for (int tap = 0; tap < 9; ++tap) loadA_c2(ac[tap], c2w, ot * 16, tap, lane);
    Frag we0, we1;
    const float* Wb = Weff + b * 64 * 64;
    loadA_mk(we0, Wb + ot * 16 * 64, 64, 0, lane);
    loadA_mk(we1, Wb + ot * 16 * 64, 64, 32, lane);
    float bias[8];
#pragma unroll
    for (int r = 0; r < 8; ++r) {
      int co = ot * 16 + r + hi * 8;
      bias[r] = c2b[co] + bp[co];
    }
    for (int row = wave >> 2; row < 16; row += 2) {
      v8f acc = {};
#pragma unroll
      for (int tap = 0; tap < 9; ++tap) {
        int dy = tap / 3, dx = tap % 3;
        Frag bf;
#pragma unroll
        for (int j = 0; j < 8; ++j)
          bf.w[j] = GB[(hi * 8 + j) * 324 + (row + dy) * 18 + n + dx];
        acc = wmmaf(ac[tap], bf, acc);
      }
      long pixg = (long)b * HW_ + (ty * 16 + row) * 256 + tx * 16 + n;
      Frag v0, v1;
#pragma unroll
      for (int j = 0; j < 8; ++j) {
        v0.w[j] = vws[(long)(hi * 8 + j) * NPIX + pixg];
        v1.w[j] = vws[(long)(16 + hi * 8 + j) * NPIX + pixg];
      }
      acc = wmmaf(we0, v0, acc);
      acc = wmmaf(we1, v1, acc);
#pragma unroll
      for (int r = 0; r < 8; r += 2)
        SB[(ot * 8 + (r >> 1) + hi * 4) * 256 + row * 16 + n] =
            pk(acc[r] + bias[r], acc[r + 1] + bias[r + 1]);
    }
  }
  __syncthreads();

  // stage 6: LayerNorm stats (mean + biased var over 64 channels per pixel)
  {
    float s = 0.f;
#pragma unroll
    for (int cp = 0; cp < 32; ++cp) {
      float2 v = up(SB[cp * 256 + t]);
      s += v.x + v.y;
    }
    float mu = s * (1.f / 64.f), vv = 0.f;
#pragma unroll
    for (int cp = 0; cp < 32; ++cp) {
      float2 v = up(SB[cp * 256 + t]);
      vv += (v.x - mu) * (v.x - mu) + (v.y - mu) * (v.y - mu);
    }
    MU[t] = mu;
    RS[t] = rsqrtf(vv * (1.f / 64.f) + 1e-6f);
  }
  __syncthreads();

  // stage 7: folded ffn_c1 (LN affine folded into W1f/b1f)
  {
    const int ot = wave & 3;
    Frag a0, a1;
    loadA_mk(a0, W1f + ot * 16 * 64, 64, 0, lane);
    loadA_mk(a1, W1f + ot * 16 * 64, 64, 32, lane);
    float bias[8];
#pragma unroll
    for (int r = 0; r < 8; ++r) bias[r] = b1f[ot * 16 + r + hi * 8];
    for (int row = wave >> 2; row < 16; row += 2) {
      int pix = row * 16 + n;
      float mu = MU[pix], rs = RS[pix];
      Frag b0, b1;
#pragma unroll
      for (int j = 0; j < 8; ++j) {
        float2 u0 = up(SB[(hi * 8 + j) * 256 + pix]);
        float2 u1 = up(SB[(16 + hi * 8 + j) * 256 + pix]);
        b0.w[j] = pk((u0.x - mu) * rs, (u0.y - mu) * rs);
        b1.w[j] = pk((u1.x - mu) * rs, (u1.y - mu) * rs);
      }
      v8f acc = {};
      acc = wmmaf(a0, b0, acc);
      acc = wmmaf(a1, b1, acc);
#pragma unroll
      for (int r = 0; r < 8; r += 2)
        HB[(ot * 8 + (r >> 1) + hi * 4) * 256 + pix] =
            pk(acc[r] + bias[r], acc[r + 1] + bias[r + 1]);
    }
  }
  __syncthreads();

  // stage 8: ffn gate
  for (int e = t; e < 16 * 256; e += 256) {
    float2 fa = up(HB[e]);
    float2 fb = up(HB[16 * 256 + e]);
    GB[e] = pk(fa.x * fb.x, fa.y * fb.y);
  }
  __syncthreads();

  // stage 9: ffn_c2 (K=32) + residuals (out + x) -> global
  {
    const int ot = wave & 3;
    Frag a0;
    loadA_mk(a0, fc2w + ot * 16 * 32, 32, 0, lane);
    float bias[8];
#pragma unroll
    for (int r = 0; r < 8; ++r) bias[r] = fc2b[ot * 16 + r + hi * 8];
    for (int row = wave >> 2; row < 16; row += 2) {
      int pix = row * 16 + n;
      Frag bf;
#pragma unroll
      for (int j = 0; j < 8; ++j) bf.w[j] = GB[(hi * 8 + j) * 256 + pix];
      v8f acc = {};
      acc = wmmaf(a0, bf, acc);
      int gy = ty * 16 + row, gxb = tx * 16;
#pragma unroll
      for (int r = 0; r < 8; ++r) {
        int co = ot * 16 + r + hi * 8;
        float ov = half_of(SB[(co >> 1) * 256 + pix], co & 1);
        float xv = half_of(XB[(co >> 1) * 400 + (row + 2) * 20 + n + 2], co & 1);
        out[(((long)b * 64 + co) << 16) + gy * 256 + gxb + n] =
            acc[r] + bias[r] + ov + xv;
      }
    }
  }
}

// ---------------- host launcher ----------------
extern "C" void kernel_launch(void* const* d_in, const int* in_sizes, int n_in,
                              void* d_out, int out_size, void* d_ws, size_t ws_size,
                              hipStream_t stream) {
  const float* x    = (const float*)d_in[0];
  const float* Wq   = (const float*)d_in[1];
  const float* Wk   = (const float*)d_in[2];
  const float* Wv   = (const float*)d_in[3];
  const float* resc = (const float*)d_in[4];
  const float* Wp   = (const float*)d_in[5];
  const float* bp   = (const float*)d_in[6];
  const float* p1w  = (const float*)d_in[7];
  const float* p1b  = (const float*)d_in[8];
  const float* dww  = (const float*)d_in[9];
  const float* dwb  = (const float*)d_in[10];
  const float* c2w  = (const float*)d_in[11];
  const float* c2b  = (const float*)d_in[12];
  const float* lnw  = (const float*)d_in[13];
  const float* lnb  = (const float*)d_in[14];
  const float* f1w  = (const float*)d_in[15];
  const float* f1b  = (const float*)d_in[16];
  const float* f2w  = (const float*)d_in[17];
  const float* f2b  = (const float*)d_in[18];
  float* outp = (float*)d_out;

  char* ws = (char*)d_ws;
  size_t off = 0;
  auto alloc = [&](size_t bytes) -> void* {
    void* p = ws + off;
    off = (off + bytes + 255) & ~(size_t)255;
    return p;
  };
  unsigned int* qws = (unsigned int*)alloc((size_t)32 * NPIX * 4);
  unsigned int* kws = (unsigned int*)alloc((size_t)32 * NPIX * 4);
  unsigned int* vws = (unsigned int*)alloc((size_t)32 * NPIX * 4);
  float* acc  = (float*)alloc(4608 * 4);          // accQ[256] accK[256] G[4096]
  float* accQ = acc, *accK = acc + 256, *G = acc + 512;
  float* Weff = (float*)alloc(4 * 64 * 64 * 4);
  float* W1f  = (float*)alloc(64 * 64 * 4);
  float* b1f  = (float*)alloc(64 * 4);

  zero_kernel<<<(4608 + 255) / 256, 256, 0, stream>>>(acc, 4608);
  qkv_kernel<<<NPIX / 256, 384, 0, stream>>>(x, Wq, Wk, Wv, qws, kws, vws);
  gram_kernel<<<dim3(HW_ / 1024, B_), 256, 0, stream>>>(qws, kws, accQ, accK, G);
  prep_kernel<<<1, 256, 0, stream>>>(accQ, accK, G, resc, Wp, lnw, lnb,
                                     f1w, f1b, Weff, W1f, b1f);
  size_t shmem = (size_t)(32 * 400 + 32 * 400 + 16 * 324 + 32 * 256) * 4 +
                 2 * 256 * 4;  // ~158 KB, fits 320 KB/WGP LDS
  fused_kernel<<<dim3(W_ / 16, H_ / 16, B_), 256, shmem, stream>>>(
      x, vws, Weff, bp, p1w, p1b, dww, dwb, c2w, c2b, W1f, b1f, f2w, f2b, outp);
}
